// STFT_69114613728935
// MI455X (gfx1250) — matrix operand order repaired and checked by reading, not compiled, not run on hardware
//
#include <hip/hip_runtime.h>
#include <hip/hip_bf16.h>

typedef float   v2f  __attribute__((ext_vector_type(2)));
typedef float   v8f  __attribute__((ext_vector_type(8)));
typedef __bf16  v16bf __attribute__((ext_vector_type(16)));
typedef unsigned short ushort16;

#define N_FFT   2048
#define HOP     512
#define PAD     1024
#define XLEN    524288
#define NFRAMES 1025
#define NFREQ   1025
#define NBATCH  16
#define WGFRAMES 32
#define WIN     (31 * HOP + N_FFT)        /* 17920 samples per WG window */
#define FTILES  65
#define TTILES  33
#define NW      (NFREQ * N_FFT)           /* 2099200 weight elements per kernel */

/* bf16 LDS row stride: 520 bf16 = 1040 B; 1040 % 256 == 16 -> b128 loads
   from 16 lanes stripe all 64 banks conflict-free. Pad 8 bf16 per 512. */
#define BSTR    520
#define LDSB    (WIN + 8 * (WIN / 512) + 8)   /* 18208 bf16 per array */

/* fp32 fallback LDS layout (stride 516 % 64 == 4 -> conflict-free b64) */
#define ROWSTR  516
#define LDSF    (WIN + 4 * (WIN / 512) + 4)

// ---------------------------------------------------------------------------
// helpers
// ---------------------------------------------------------------------------
__device__ __forceinline__ unsigned short f32_to_bf16_rne(float f) {
    unsigned int u = __float_as_uint(f);
    unsigned int r = u + 0x7FFFu + ((u >> 16) & 1u);
    return (unsigned short)(r >> 16);
}
__device__ __forceinline__ float bf16_bits_to_f32(unsigned short h) {
    return __uint_as_float(((unsigned int)h) << 16);
}

union Frag {                 // 16 bf16 = 8 VGPRs, loaded as two b128 chunks
    v16bf v;
    uint4 q[2];
};

__device__ __forceinline__ v8f wmma_bf16(const Frag& A, const Frag& B, v8f C) {
    return __builtin_amdgcn_wmma_f32_16x16x32_bf16(
        false, A.v, false, B.v, (short)0, C, false, false);
}

// ---------------------------------------------------------------------------
// Prep: split fp32 weights into bf16 hi/lo (RNE), row-major [1025][2048]
// ---------------------------------------------------------------------------
__global__ __launch_bounds__(256)
void split_weights_kernel(const float* __restrict__ wsin,
                          const float* __restrict__ wcos,
                          unsigned short* __restrict__ cos_hi,
                          unsigned short* __restrict__ cos_lo,
                          unsigned short* __restrict__ sin_hi,
                          unsigned short* __restrict__ sin_lo)
{
    for (int i = blockIdx.x * blockDim.x + threadIdx.x; i < NW;
         i += gridDim.x * blockDim.x) {
        float c = wcos[i];
        unsigned short ch = f32_to_bf16_rne(c);
        cos_hi[i] = ch;
        cos_lo[i] = f32_to_bf16_rne(c - bf16_bits_to_f32(ch));
        float s = wsin[i];
        unsigned short sh = f32_to_bf16_rne(s);
        sin_hi[i] = sh;
        sin_lo[i] = f32_to_bf16_rne(s - bf16_bits_to_f32(sh));
    }
}

// ---------------------------------------------------------------------------
// Main: bf16x3 split GEMM on v_wmma_f32_16x16x32_bf16
// ---------------------------------------------------------------------------
__global__ __launch_bounds__(128)
void stft_wmma_bf16x3_kernel(const float* __restrict__ x,
                             const unsigned short* __restrict__ cos_hi,
                             const unsigned short* __restrict__ cos_lo,
                             const unsigned short* __restrict__ sin_hi,
                             const unsigned short* __restrict__ sin_lo,
                             float* __restrict__ out)
{
    __shared__ __align__(16) unsigned short xh[LDSB];
    __shared__ __align__(16) unsigned short xl[LDSB];

    const int tile_t = blockIdx.x;
    const int b      = blockIdx.y;
    const int tid    = threadIdx.x;
    const int lane   = tid & 31;
    const int wave   = tid >> 5;
    const int llo    = lane & 15;
    const int lhi    = lane >> 4;

    const float* xb = x + (size_t)b * XLEN;
    const long w_start = (long)tile_t * (WGFRAMES * HOP) - PAD;

    // ---- Stage window, splitting to bf16 hi/lo; pad 8 bf16 every 512 ----
    for (int i = tid * 8; i < WIN; i += 128 * 8) {
        float s[8];
        long g = w_start + i;
        if (g >= 0 && g + 7 < (long)XLEN) {
            float4 v0 = *reinterpret_cast<const float4*>(xb + g);
            float4 v1 = *reinterpret_cast<const float4*>(xb + g + 4);
            s[0]=v0.x; s[1]=v0.y; s[2]=v0.z; s[3]=v0.w;
            s[4]=v1.x; s[5]=v1.y; s[6]=v1.z; s[7]=v1.w;
        } else {
#pragma unroll
            for (int j = 0; j < 8; ++j) {
                long gj = g + j;
                s[j] = (gj >= 0 && gj < (long)XLEN) ? xb[gj] : 0.0f;
            }
        }
        unsigned int H[4], L[4];
#pragma unroll
        for (int j = 0; j < 4; ++j) {
            unsigned short h0 = f32_to_bf16_rne(s[2*j]);
            unsigned short h1 = f32_to_bf16_rne(s[2*j+1]);
            unsigned short l0 = f32_to_bf16_rne(s[2*j]   - bf16_bits_to_f32(h0));
            unsigned short l1 = f32_to_bf16_rne(s[2*j+1] - bf16_bits_to_f32(h1));
            H[j] = (unsigned int)h0 | ((unsigned int)h1 << 16);
            L[j] = (unsigned int)l0 | ((unsigned int)l1 << 16);
        }
        int d = i + ((i >> 9) << 3);   // +8 bf16 pad per 512 samples
        *reinterpret_cast<uint4*>(&xh[d]) = make_uint4(H[0], H[1], H[2], H[3]);
        *reinterpret_cast<uint4*>(&xl[d]) = make_uint4(L[0], L[1], L[2], L[3]);
    }
    __syncthreads();

    const int t0 = tile_t * WGFRAMES + llo;
    const int t1 = t0 + 16;

    for (int ftile = wave; ftile < FTILES; ftile += 4) {
        const int f0   = ftile * 16;
        const int arow = min(f0 + llo, NFREQ - 1);
        const size_t rbase = (size_t)arow * N_FFT + 16 * lhi;  // A chunk0 base
        const unsigned short* pch = cos_hi + rbase;
        const unsigned short* pcl = cos_lo + rbase;
        const unsigned short* psh = sin_hi + rbase;
        const unsigned short* psl = sin_lo + rbase;

        v8f ar0 = {0.f,0.f,0.f,0.f,0.f,0.f,0.f,0.f};
        v8f ai0 = {0.f,0.f,0.f,0.f,0.f,0.f,0.f,0.f};
        v8f ar1 = {0.f,0.f,0.f,0.f,0.f,0.f,0.f,0.f};
        v8f ai1 = {0.f,0.f,0.f,0.f,0.f,0.f,0.f,0.f};

        for (int kb = 0; kb < 4; ++kb) {
            const int kw = kb * 512;                 // weight K base
            const int bbase = (llo + kb) * BSTR + 16 * lhi;  // B K-run base in LDS

#pragma unroll 2
            for (int kk = 0; kk < 512; kk += 32) {
                // A fragments: 16 consecutive K at k0+16*lhi
                Frag ch, cl, sh, sl;
                ch.q[0] = *reinterpret_cast<const uint4*>(pch + kw + kk);
                ch.q[1] = *reinterpret_cast<const uint4*>(pch + kw + kk + 8);
                cl.q[0] = *reinterpret_cast<const uint4*>(pcl + kw + kk);
                cl.q[1] = *reinterpret_cast<const uint4*>(pcl + kw + kk + 8);
                sh.q[0] = *reinterpret_cast<const uint4*>(psh + kw + kk);
                sh.q[1] = *reinterpret_cast<const uint4*>(psh + kw + kk + 8);
                sl.q[0] = *reinterpret_cast<const uint4*>(psl + kw + kk);
                sl.q[1] = *reinterpret_cast<const uint4*>(psl + kw + kk + 8);
                // B fragments: 16 consecutive K at k0+16*lhi (two b128s)
                Frag bh0, bl0, bh1, bl1;
                bh0.q[0] = *reinterpret_cast<const uint4*>(&xh[bbase + kk]);
                bh0.q[1] = *reinterpret_cast<const uint4*>(&xh[bbase + kk + 8]);
                bl0.q[0] = *reinterpret_cast<const uint4*>(&xl[bbase + kk]);
                bl0.q[1] = *reinterpret_cast<const uint4*>(&xl[bbase + kk + 8]);
                bh1.q[0] = *reinterpret_cast<const uint4*>(&xh[bbase + 16*BSTR + kk]);
                bh1.q[1] = *reinterpret_cast<const uint4*>(&xh[bbase + 16*BSTR + kk + 8]);
                bl1.q[0] = *reinterpret_cast<const uint4*>(&xl[bbase + 16*BSTR + kk]);
                bl1.q[1] = *reinterpret_cast<const uint4*>(&xl[bbase + 16*BSTR + kk + 8]);

                // bf16x3: hi*hi + hi*lo + lo*hi, f32 accumulate
                ar0 = wmma_bf16(ch, bh0, ar0);
                ar0 = wmma_bf16(ch, bl0, ar0);
                ar0 = wmma_bf16(cl, bh0, ar0);
                ai0 = wmma_bf16(sh, bh0, ai0);
                ai0 = wmma_bf16(sh, bl0, ai0);
                ai0 = wmma_bf16(sl, bh0, ai0);
                ar1 = wmma_bf16(ch, bh1, ar1);
                ar1 = wmma_bf16(ch, bl1, ar1);
                ar1 = wmma_bf16(cl, bh1, ar1);
                ai1 = wmma_bf16(sh, bh1, ai1);
                ai1 = wmma_bf16(sh, bl1, ai1);
                ai1 = wmma_bf16(sl, bh1, ai1);
            }
        }

#pragma unroll
        for (int r = 0; r < 8; ++r) {
            const int f = f0 + r + 8 * lhi;
            const bool fok = (f < NFREQ);
            {
                float re = ar0[r], im = ai0[r];
                float val = 10.0f * __log10f(re * re + im * im);
                if (fok && t0 < NFRAMES)
                    out[((size_t)b * NFREQ + f) * (size_t)NFRAMES + t0] = val;
            }
            {
                float re = ar1[r], im = ai1[r];
                float val = 10.0f * __log10f(re * re + im * im);
                if (fok && t1 < NFRAMES)
                    out[((size_t)b * NFREQ + f) * (size_t)NFRAMES + t1] = val;
            }
        }
    }
}

// ---------------------------------------------------------------------------
// Fallback: pure fp32 WMMA 16x16x4 (used when d_ws too small)
// ---------------------------------------------------------------------------
__global__ __launch_bounds__(128)
void stft_wmma_f32_kernel(const float* __restrict__ x,
                          const float* __restrict__ wsin,
                          const float* __restrict__ wcos,
                          float* __restrict__ out)
{
    __shared__ float xs[LDSF];

    const int tile_t = blockIdx.x;
    const int b      = blockIdx.y;
    const int tid    = threadIdx.x;
    const int lane   = tid & 31;
    const int wave   = tid >> 5;
    const int llo    = lane & 15;
    const int lhi    = lane >> 4;

    const float* xb = x + (size_t)b * XLEN;
    const long w_start = (long)tile_t * (WGFRAMES * HOP) - PAD;

    for (int i = tid * 4; i < WIN; i += 128 * 4) {
        long g = w_start + i;
        float4 v;
        if (g >= 0 && g + 3 < (long)XLEN) {
            v = *reinterpret_cast<const float4*>(xb + g);
        } else {
            float tmp[4];
#pragma unroll
            for (int j = 0; j < 4; ++j) {
                long gj = g + j;
                tmp[j] = (gj >= 0 && gj < (long)XLEN) ? xb[gj] : 0.0f;
            }
            v = make_float4(tmp[0], tmp[1], tmp[2], tmp[3]);
        }
        int d = i + ((i >> 9) << 2);
        *reinterpret_cast<float4*>(&xs[d]) = v;
    }
    __syncthreads();

    const int t0 = tile_t * WGFRAMES + llo;
    const int t1 = t0 + 16;

    for (int ftile = wave; ftile < FTILES; ftile += 4) {
        const int f0   = ftile * 16;
        const int arow = min(f0 + llo, NFREQ - 1);
        const float* wc = wcos + (size_t)arow * N_FFT + 2 * lhi;
        const float* ws = wsin + (size_t)arow * N_FFT + 2 * lhi;

        v8f ar0 = {0.f,0.f,0.f,0.f,0.f,0.f,0.f,0.f};
        v8f ai0 = {0.f,0.f,0.f,0.f,0.f,0.f,0.f,0.f};
        v8f ar1 = {0.f,0.f,0.f,0.f,0.f,0.f,0.f,0.f};
        v8f ai1 = {0.f,0.f,0.f,0.f,0.f,0.f,0.f,0.f};

        for (int kb = 0; kb < 4; ++kb) {
            const float* pc = wc + kb * 512;
            const float* ps = ws + kb * 512;
            const float* p0 = &xs[(llo + kb) * ROWSTR + 2 * lhi];
            const float* p1 = p0 + 16 * ROWSTR;

#pragma unroll 4
            for (int kk = 0; kk < 512; kk += 4) {
                v2f ac = *reinterpret_cast<const v2f*>(pc + kk);
                v2f as = *reinterpret_cast<const v2f*>(ps + kk);
                v2f b0 = *reinterpret_cast<const v2f*>(p0 + kk);
                v2f b1 = *reinterpret_cast<const v2f*>(p1 + kk);
                ar0 = __builtin_amdgcn_wmma_f32_16x16x4_f32(false, ac, false, b0, (short)0, ar0, false, false);
                ai0 = __builtin_amdgcn_wmma_f32_16x16x4_f32(false, as, false, b0, (short)0, ai0, false, false);
                ar1 = __builtin_amdgcn_wmma_f32_16x16x4_f32(false, ac, false, b1, (short)0, ar1, false, false);
                ai1 = __builtin_amdgcn_wmma_f32_16x16x4_f32(false, as, false, b1, (short)0, ai1, false, false);
            }
        }

#pragma unroll
        for (int r = 0; r < 8; ++r) {
            const int f = f0 + r + 8 * lhi;
            const bool fok = (f < NFREQ);
            {
                float re = ar0[r], im = ai0[r];
                float val = 10.0f * __log10f(re * re + im * im);
                if (fok && t0 < NFRAMES)
                    out[((size_t)b * NFREQ + f) * (size_t)NFRAMES + t0] = val;
            }
            {
                float re = ar1[r], im = ai1[r];
                float val = 10.0f * __log10f(re * re + im * im);
                if (fok && t1 < NFRAMES)
                    out[((size_t)b * NFREQ + f) * (size_t)NFRAMES + t1] = val;
            }
        }
    }
}

// ---------------------------------------------------------------------------
extern "C" void kernel_launch(void* const* d_in, const int* in_sizes, int n_in,
                              void* d_out, int out_size, void* d_ws, size_t ws_size,
                              hipStream_t stream) {
    const float* x    = (const float*)d_in[0];
    const float* wsin = (const float*)d_in[1];
    const float* wcos = (const float*)d_in[2];
    float* out        = (float*)d_out;

    const size_t arr  = (size_t)NW * sizeof(unsigned short);  // 4,198,400 B
    const size_t need = 4 * arr;                              // 16.8 MB

    dim3 grid(TTILES, NBATCH);

    if (ws_size >= need) {
        unsigned short* cos_hi = (unsigned short*)d_ws;
        unsigned short* cos_lo = cos_hi + NW;
        unsigned short* sin_hi = cos_lo + NW;
        unsigned short* sin_lo = sin_hi + NW;
        split_weights_kernel<<<2048, 256, 0, stream>>>(
            wsin, wcos, cos_hi, cos_lo, sin_hi, sin_lo);
        stft_wmma_bf16x3_kernel<<<grid, 128, 0, stream>>>(
            x, cos_hi, cos_lo, sin_hi, sin_lo, out);
    } else {
        stft_wmma_f32_kernel<<<grid, 128, 0, stream>>>(x, wsin, wcos, out);
    }
}
